// ResAttention_70909910057459
// MI455X (gfx1250) — compile-verified, matrix operand order checked by reading, not run on hardware
//
#include <hip/hip_runtime.h>
#include <hip/hip_bf16.h>

// ---------------------------------------------------------------------------
// Residual multi-head attention block for MI455X (gfx1250, wave32, WMMA).
//   qp = q@Wq.T, kp = k@Wk.T, vp = k@Wv.T          (WMMA f16 GEMMs, f32 accum)
//   per head-batch unit: S = (Q Khᵀ)/32, mask==0 -> 0, softmax, O = P Vh
//   mha = o@Wo.T ; out = LayerNorm(q + mha)
// ---------------------------------------------------------------------------

typedef __attribute__((ext_vector_type(16))) _Float16 v16h;
typedef __attribute__((ext_vector_type(8)))  _Float16 v8h;
typedef __attribute__((ext_vector_type(4)))  _Float16 v4h;
typedef __attribute__((ext_vector_type(8)))  float    v8f;

static constexpr int Bc  = 4;
static constexpr int Sc  = 1024;
static constexpr int Dc  = 1024;   // DIM
static constexpr int Hc  = 16;
static constexpr int DHc = 64;
static constexpr int Mc  = Bc * Sc;   // 4096 rows in GEMMs

// LDS row strides (padded to dodge 64-bank conflicts; see analysis).
static constexpr int SSs = 1028;   // f32 score row stride (floats)
static constexpr int SPs = 1032;   // f16 prob row stride (halves, 16B-aligned)

// Load a 16x32 (A) or 32x16-col-major (B) f16 WMMA fragment from K-contiguous
// storage. Lane l (r=l&15,g=l>>4) holds K = g*8+j and 16+g*8+j of row/col r.
// Caller passes p already offset by (r*ld + g*8); 16B-aligned.
__device__ __forceinline__ v16h ldfrag(const _Float16* p) {
  v8h lo = *(const v8h*)(p);
  v8h hi = *(const v8h*)(p + 16);
  return __builtin_shufflevector(lo, hi, 0,1,2,3,4,5,6,7,8,9,10,11,12,13,14,15);
}

__device__ __forceinline__ v8f wmma16(v16h a, v16h b, v8f c) {
  return __builtin_amdgcn_wmma_f32_16x16x32_f16(
      /*neg_a=*/false, a, /*neg_b=*/false, b,
      /*c_mod=*/(short)0, c, /*reuse_a=*/false, /*reuse_b=*/false);
}

// ---------------------------------------------------------------------------
__global__ void cvt_f32_to_f16(const float* __restrict__ src,
                               _Float16* __restrict__ dst, int n4) {
  int i = blockIdx.x * blockDim.x + threadIdx.x;
  if (i < n4) {
    float4 f = reinterpret_cast<const float4*>(src)[i];
    v4h h;
    h[0] = (_Float16)f.x; h[1] = (_Float16)f.y;
    h[2] = (_Float16)f.z; h[3] = (_Float16)f.w;
    reinterpret_cast<v4h*>(dst)[i] = h;
  }
}

__global__ void zero_int_kernel(int* p) { *p = 0; }

__global__ void mask_sum_kernel(const int* __restrict__ mask, int n,
                                int* __restrict__ out) {
  int acc = 0;
  for (int i = blockIdx.x * blockDim.x + threadIdx.x; i < n;
       i += gridDim.x * blockDim.x)
    acc += mask[i];
  for (int off = 16; off; off >>= 1) acc += __shfl_xor(acc, off, 32);
  if ((threadIdx.x & 31) == 0) atomicAdd(out, acc);
}

// ---------------------------------------------------------------------------
// Unified GEMM: C[M,1024] = A[M,1024](f16) x Bw[1024,1024](f16, N-by-K rows).
// Block 256 thr = 8 waves (2 in M x 4 in N); wave tile 32x64 (8 WMMA accums,
// 6 fragment loads / 8 WMMAs per k-step); block tile 64(M) x 256(N).
// mode 0: half out [M,N]; mode 1: half out transposed into vT[b][n][s];
// mode 2: f32 out [M,N].
__global__ __launch_bounds__(256) void gemm_wmma_kernel(
    const _Float16* __restrict__ A, const _Float16* __restrict__ Bw,
    _Float16* __restrict__ Ch, float* __restrict__ Cf, int mode) {
  constexpr int K = Dc, N = Dc;
  const int lane = threadIdx.x & 31;
  const int wave = threadIdx.x >> 5;
  const int wm = wave & 1, wn = wave >> 1;
  const int r = lane & 15, g = lane >> 4;
  const int mBase = blockIdx.y * 64 + wm * 32;
  const int nBase = blockIdx.x * 256 + wn * 64;

  const _Float16* aRow[2];
  const _Float16* bRow[4];
#pragma unroll
  for (int mi = 0; mi < 2; ++mi)
    aRow[mi] = A + (size_t)(mBase + mi * 16 + r) * K + g * 8;
#pragma unroll
  for (int ni = 0; ni < 4; ++ni)
    bRow[ni] = Bw + (size_t)(nBase + ni * 16 + r) * K + g * 8;

  v8f acc[2][4] = {};
  for (int k = 0; k < K; k += 32) {
    v16h a0 = ldfrag(aRow[0] + k);
    v16h a1 = ldfrag(aRow[1] + k);
#pragma unroll
    for (int ni = 0; ni < 4; ++ni) {
      v16h b = ldfrag(bRow[ni] + k);
      acc[0][ni] = wmma16(a0, b, acc[0][ni]);
      acc[1][ni] = wmma16(a1, b, acc[1][ni]);
    }
  }

#pragma unroll
  for (int mi = 0; mi < 2; ++mi)
#pragma unroll
    for (int ni = 0; ni < 4; ++ni) {
      const int mg = mBase + mi * 16 + g * 8;     // +i rows below
      const int ng = nBase + ni * 16 + r;         // lane's column
      const v8f c = acc[mi][ni];
      if (mode == 0) {
#pragma unroll
        for (int i = 0; i < 8; ++i)
          Ch[(size_t)(mg + i) * N + ng] = (_Float16)c[i];
      } else if (mode == 1) {                     // vT[b][ng][s], s contiguous
        const int bb = mg >> 10, ss = mg & 1023;
        _Float16* p = Ch + ((size_t)bb * Dc + ng) * Sc + ss;
#pragma unroll
        for (int i = 0; i < 8; ++i) p[i] = (_Float16)c[i];
      } else {
#pragma unroll
        for (int i = 0; i < 8; ++i)
          Cf[(size_t)(mg + i) * N + ng] = c[i];
      }
    }
}

// ---------------------------------------------------------------------------
// Fused attention: grid (S/32 query blocks, 64 units). Unit n: h=n/4, b=n%4,
// mask batch mb=n/16 (faithful repeat_interleave mismatch). Dynamic LDS:
// 32x1028 f32 scores + 32x1032 f16 probs = 193 KB (< 320 KB/WGP).
__global__ __launch_bounds__(256) void attn_kernel(
    const _Float16* __restrict__ Qp, const _Float16* __restrict__ Kp,
    const _Float16* __restrict__ Vt, const int* __restrict__ mask,
    const int* __restrict__ masksum, _Float16* __restrict__ Oh) {
  extern __shared__ char smem[];
  float*    s_s = (float*)smem;                            // [32][SSs]
  _Float16* s_p = (_Float16*)(smem + 32 * SSs * 4);        // [32][SPs]

  const int n = blockIdx.y;
  const int h = n >> 2, b = n & 3, mb = n >> 4;
  const int q0 = blockIdx.x * 32;
  const int lane = threadIdx.x & 31, wave = threadIdx.x >> 5;
  const int r = lane & 15, g = lane >> 4;

  const _Float16* Qb = Qp + (size_t)b * Sc * Dc + h * DHc;
  const _Float16* Kb = Kp + (size_t)b * Sc * Dc + h * DHc;

  // Q fragments: 2 M-tiles x 2 K-steps (K = DH = 64), reused for all keys.
  v16h qf[2][2];
#pragma unroll
  for (int mt = 0; mt < 2; ++mt)
#pragma unroll
    for (int ks = 0; ks < 2; ++ks)
      qf[mt][ks] = ldfrag(Qb + (size_t)(q0 + mt * 16 + r) * Dc + ks * 32 + g * 8);

  // Scores: each wave owns a 128-key strip (8 N-tiles of 16).
  for (int nt = 0; nt < 8; ++nt) {
    const int kcol = wave * 128 + nt * 16;
    v8f c0 = {}, c1 = {};
#pragma unroll
    for (int ks = 0; ks < 2; ++ks) {
      v16h kf = ldfrag(Kb + (size_t)(kcol + r) * Dc + ks * 32 + g * 8);
      c0 = wmma16(qf[0][ks], kf, c0);
      c1 = wmma16(qf[1][ks], kf, c1);
    }
#pragma unroll
    for (int i = 0; i < 8; ++i) {
      s_s[(g * 8 + i) * SSs + kcol + r]      = c0[i];
      s_s[(16 + g * 8 + i) * SSs + kcol + r] = c1[i];
    }
  }
  __syncthreads();

  // Scale by 1/sqrt(DIM)=1/32, then masked_fill to ZERO (reference quirk).
  const int* mrow = mask + ((size_t)mb * Sc + q0) * Sc;
  for (int idx = threadIdx.x; idx < 32 * 1024; idx += 256) {
    const int qq = idx >> 10, kk = idx & 1023;
    float v = s_s[qq * SSs + kk] * 0.03125f;
    s_s[qq * SSs + kk] = (mrow[(size_t)qq * Sc + kk] == 0) ? 0.0f : v;
  }
  __syncthreads();

  // Softmax over all 1024 keys (masked zeros participate, as in reference).
  for (int rr = 0; rr < 4; ++rr) {
    const int q = wave * 4 + rr;
    float mx = -1e30f;
    for (int kk = lane; kk < 1024; kk += 32) mx = fmaxf(mx, s_s[q * SSs + kk]);
    for (int off = 16; off; off >>= 1) mx = fmaxf(mx, __shfl_xor(mx, off, 32));
    float sum = 0.f;
    for (int kk = lane; kk < 1024; kk += 32) {
      float e = __expf(s_s[q * SSs + kk] - mx);
      s_s[q * SSs + kk] = e;
      sum += e;
    }
    for (int off = 16; off; off >>= 1) sum += __shfl_xor(sum, off, 32);
    const float rinv = 1.0f / sum;
    for (int kk = lane; kk < 1024; kk += 32)
      s_p[q * SPs + kk] = (_Float16)(s_s[q * SSs + kk] * rinv);
  }
  __syncthreads();

  // O = P(32x1024) x V(1024x64): 8 C-tiles, one per wave. K loop 32 steps.
  const int mt = wave & 1, nt = wave >> 1;
  const _Float16* Vb = Vt + ((size_t)b * Dc + h * DHc + nt * 16) * Sc;
  v8f acc = {};
  for (int ks = 0; ks < 32; ++ks) {
    v16h pa = ldfrag(s_p + (size_t)(mt * 16 + r) * SPs + ks * 32 + g * 8);
    v16h vb = ldfrag(Vb + (size_t)r * Sc + ks * 32 + g * 8);
    acc = wmma16(pa, vb, acc);
  }
  const float flag = (*masksum > 0) ? 1.0f : 0.0f;  // global all-zero branch
  _Float16* Ob = Oh + ((size_t)b * Sc + q0 + mt * 16) * Dc + h * DHc + nt * 16;
#pragma unroll
  for (int i = 0; i < 8; ++i)
    Ob[(size_t)(g * 8 + i) * Dc + r] = (_Float16)(acc[i] * flag);
}

// ---------------------------------------------------------------------------
__global__ __launch_bounds__(256) void add_ln_kernel(
    const float* __restrict__ qin, const float* __restrict__ mha,
    const float* __restrict__ gamma, const float* __restrict__ beta,
    float* __restrict__ out) {
  const int row = blockIdx.x;
  const float* xq = qin + (size_t)row * Dc;
  const float* xm = mha + (size_t)row * Dc;
  float x[4], s = 0.f, s2 = 0.f;
#pragma unroll
  for (int j = 0; j < 4; ++j) {
    const int c = threadIdx.x + j * 256;
    x[j] = xq[c] + xm[c];
    s += x[j]; s2 += x[j] * x[j];
  }
  for (int off = 16; off; off >>= 1) {
    s  += __shfl_xor(s,  off, 32);
    s2 += __shfl_xor(s2, off, 32);
  }
  __shared__ float rs[8], rs2[8];
  const int wave = threadIdx.x >> 5, lane = threadIdx.x & 31;
  if (lane == 0) { rs[wave] = s; rs2[wave] = s2; }
  __syncthreads();
  float tot = 0.f, tot2 = 0.f;
#pragma unroll
  for (int i = 0; i < 8; ++i) { tot += rs[i]; tot2 += rs2[i]; }
  const float mean = tot * (1.0f / 1024.0f);
  const float var  = tot2 * (1.0f / 1024.0f) - mean * mean;
  const float rstd = rsqrtf(var + 1e-5f);
#pragma unroll
  for (int j = 0; j < 4; ++j) {
    const int c = threadIdx.x + j * 256;
    out[(size_t)row * Dc + c] = gamma[c] * (x[j] - mean) * rstd + beta[c];
  }
}

// ---------------------------------------------------------------------------
extern "C" void kernel_launch(void* const* d_in, const int* in_sizes, int n_in,
                              void* d_out, int out_size, void* d_ws,
                              size_t ws_size, hipStream_t stream) {
  (void)in_sizes; (void)n_in; (void)out_size; (void)ws_size;
  const float* q    = (const float*)d_in[0];
  const float* k    = (const float*)d_in[1];
  const int*   mask = (const int*)d_in[2];
  const float* Wq   = (const float*)d_in[3];
  const float* Wk   = (const float*)d_in[4];
  const float* Wv   = (const float*)d_in[5];
  const float* Wo   = (const float*)d_in[6];
  const float* ga   = (const float*)d_in[7];
  const float* be   = (const float*)d_in[8];
  float* out = (float*)d_out;

  char* ws = (char*)d_ws;
  size_t off = 0;
  auto alloc = [&](size_t bytes) -> void* {
    void* p = ws + off;
    off += (bytes + 255) & ~(size_t)255;
    return p;
  };
  const size_t QK_BYTES = (size_t)Mc * Dc * sizeof(_Float16);  // 8 MB
  const size_t W_BYTES  = (size_t)Dc * Dc * sizeof(_Float16);  // 2 MB
  _Float16* qh   = (_Float16*)alloc(QK_BYTES);
  _Float16* kh   = (_Float16*)alloc(QK_BYTES);
  _Float16* Wq16 = (_Float16*)alloc(W_BYTES);
  _Float16* Wk16 = (_Float16*)alloc(W_BYTES);
  _Float16* Wv16 = (_Float16*)alloc(W_BYTES);
  _Float16* Wo16 = (_Float16*)alloc(W_BYTES);
  _Float16* qp16 = (_Float16*)alloc(QK_BYTES);
  _Float16* kp16 = (_Float16*)alloc(QK_BYTES);
  _Float16* vT16 = (_Float16*)alloc(QK_BYTES);
  _Float16* oh16 = (_Float16*)alloc(QK_BYTES);
  float*    mha  = (float*)alloc((size_t)Mc * Dc * sizeof(float));
  int*      flag = (int*)alloc(sizeof(int));

  auto cvt = [&](const float* src, _Float16* dst, int n) {
    const int n4 = n / 4;
    cvt_f32_to_f16<<<(n4 + 255) / 256, 256, 0, stream>>>(src, dst, n4);
  };
  cvt(q,  qh,   Mc * Dc);
  cvt(k,  kh,   Mc * Dc);
  cvt(Wq, Wq16, Dc * Dc);
  cvt(Wk, Wk16, Dc * Dc);
  cvt(Wv, Wv16, Dc * Dc);
  cvt(Wo, Wo16, Dc * Dc);

  zero_int_kernel<<<1, 1, 0, stream>>>(flag);
  mask_sum_kernel<<<1024, 256, 0, stream>>>(mask, Bc * Sc * Sc, flag);

  const dim3 ggrid(Dc / 256, Mc / 64);
  gemm_wmma_kernel<<<ggrid, 256, 0, stream>>>(qh, Wq16, qp16, nullptr, 0);
  gemm_wmma_kernel<<<ggrid, 256, 0, stream>>>(kh, Wk16, kp16, nullptr, 0);
  gemm_wmma_kernel<<<ggrid, 256, 0, stream>>>(kh, Wv16, vT16, nullptr, 1);

  const size_t attn_lds = 32 * SSs * 4 + 32 * SPs * 2;  // ~193 KB
  (void)hipFuncSetAttribute((const void*)attn_kernel,
                            hipFuncAttributeMaxDynamicSharedMemorySize,
                            (int)attn_lds);
  attn_kernel<<<dim3(Sc / 32, Hc * Bc), 256, attn_lds, stream>>>(
      qp16, kp16, vT16, mask, flag, oh16);

  gemm_wmma_kernel<<<ggrid, 256, 0, stream>>>(oh16, Wo16, nullptr, mha, 2);

  add_ln_kernel<<<Mc, 256, 0, stream>>>(q, mha, ga, be, out);
}
